// ExplicitMultiheadAttention_14989435863259
// MI455X (gfx1250) — matrix operand order repaired and checked by reading, not compiled, not run on hardware
//
#include <hip/hip_runtime.h>
#include <hip/hip_bf16.h>

// ---------------------------------------------------------------------------
// Types for CDNA5 WMMA (wave32): 16x16x32 bf16, f32 accumulate
// ---------------------------------------------------------------------------
typedef __bf16 bf16_t;
typedef __attribute__((ext_vector_type(8)))  __bf16 v8bf;
typedef __attribute__((ext_vector_type(16))) __bf16 v16bf;
typedef __attribute__((ext_vector_type(8)))  float  v8f;

#define EMBED 1024
#define HEADS 16
#define HDIM  64
#define SEQ   2048
#define BATCH 2
#define MROWS (BATCH * SEQ)           // 4096
#define BSE   ((size_t)MROWS * EMBED) // 4,194,304 elements
#define EE    ((size_t)EMBED * EMBED) // 1,048,576 elements

// ---------------------------------------------------------------------------
// gfx1250 async global->LDS copy (ASYNCcnt path), with safe fallbacks
// ---------------------------------------------------------------------------
#if defined(__AMDGCN__) && __has_builtin(__builtin_amdgcn_global_load_async_to_lds_b128)
#define HAVE_ASYNC_LDS 1
#else
#define HAVE_ASYNC_LDS 0
#endif

#if HAVE_ASYNC_LDS
typedef __attribute__((__vector_size__(4 * sizeof(int)))) int v4i_t;
typedef __attribute__((address_space(1))) v4i_t gv4i_t;   // global
typedef __attribute__((address_space(3))) v4i_t lv4i_t;   // LDS
#endif

__device__ __forceinline__ void async_copy16(const void* g, void* l) {
#if HAVE_ASYNC_LDS
  __builtin_amdgcn_global_load_async_to_lds_b128((gv4i_t*)g, (lv4i_t*)l, 0, 0);
#else
  *(float4*)l = *(const float4*)g;   // sync fallback: load + ds_store
#endif
}

#if HAVE_ASYNC_LDS
#if __has_builtin(__builtin_amdgcn_s_wait_asynccnt)
#define WAIT_ASYNC(n) __builtin_amdgcn_s_wait_asynccnt(n)
#else
#define WAIT_ASYNC(n) asm volatile("s_wait_asynccnt %0" ::"i"(n) : "memory")
#endif
#else
#define WAIT_ASYNC(n) ((void)0)
#endif

// ---------------------------------------------------------------------------
// WMMA helpers
// ---------------------------------------------------------------------------
__device__ __forceinline__ v8f wmma_bf16(v16bf a, v16bf b, v8f c) {
  // (neg_a, A, neg_b, B, c_mod, C, reuse_a, reuse_b)
  return __builtin_amdgcn_wmma_f32_16x16x32_bf16(false, a, false, b,
                                                 (short)0, c, false, false);
}

__device__ __forceinline__ v16bf cat16(v8bf lo, v8bf hi) {
  return __builtin_shufflevector(lo, hi, 0, 1, 2, 3, 4, 5, 6, 7,
                                 8, 9, 10, 11, 12, 13, 14, 15);
}

// A fragment (16 rows x 32 K, bf16, row-major, leading dim ld elements).
// ISA 16-bit A layout: lanes0-15 -> M=lane, K chunks [0..7]+[16..23];
// lanes16-31 -> M=lane-16, K chunks [8..15]+[24..31].
__device__ __forceinline__ v16bf load_A(const bf16_t* src, int ld, int lane) {
  int m  = lane & 15;
  int kb = (lane < 16) ? 0 : 8;
  const bf16_t* p = src + (size_t)m * ld + kb;
  v8bf lo = *(const v8bf*)(p);
  v8bf hi = *(const v8bf*)(p + 16);
  return cat16(lo, hi);
}

// B fragment (32 K x 16 N, bf16, K-major): N = lane&15; K chunks [0..7]+[16..23]
// for lanes0-15, [8..15]+[24..31] for lanes16-31 (matches A).
__device__ __forceinline__ v16bf load_B(const bf16_t* src, int ld, int lane) {
  int n  = lane & 15;
  int kb = (lane < 16) ? 0 : 8;
  v16bf r;
#pragma unroll
  for (int s = 0; s < 8; ++s) {
    r[s]     = src[(size_t)(kb + s) * ld + n];
    r[s + 8] = src[(size_t)(16 + kb + s) * ld + n];
  }
  return r;
}

// ---------------------------------------------------------------------------
// Cast helpers
// ---------------------------------------------------------------------------
__global__ __launch_bounds__(256) void cast_f32_to_bf16(
    const float* __restrict__ in, bf16_t* __restrict__ out, size_t n) {
  size_t i = (size_t)blockIdx.x * blockDim.x + threadIdx.x;
  if (i < n) out[i] = (bf16_t)in[i];
}

// W [N][K] f32 (torch Linear weight) -> Wt [K][N] bf16, LDS tile transpose.
__global__ __launch_bounds__(256) void transpose_cast_w(
    const float* __restrict__ W, bf16_t* __restrict__ Wt) {
  __shared__ float tile[32][33];
  int bk = blockIdx.x * 32;   // k block
  int bn = blockIdx.y * 32;   // n block
  int tx = threadIdx.x & 31;  // fast dim
  int ty = threadIdx.x >> 5;  // 0..7
#pragma unroll
  for (int i = 0; i < 4; ++i) {
    int n = bn + ty + 8 * i;
    tile[ty + 8 * i][tx] = W[(size_t)n * EMBED + bk + tx]; // coalesced read
  }
  __syncthreads();
#pragma unroll
  for (int i = 0; i < 4; ++i) {
    int k = bk + ty + 8 * i;
    Wt[(size_t)k * EMBED + bn + tx] = (bf16_t)tile[tx][ty + 8 * i]; // coalesced write
  }
}

// ---------------------------------------------------------------------------
// WMMA GEMM: Y[M,N] = X[M,K] @ Wt[K,N] + bias[N]
// Double-buffered async global->LDS staging of both tiles:
//   X tile 64x32 bf16 (padded ld=40), W tile 32x128 bf16 (padded ld=136).
// Block: 256 threads = 8 waves; tile 64(M) x 128(N); wave tile 16 x 64.
// mode 0: fp32 out [M,N]; mode 1: bf16 [B,H,S,D]; mode 2: bf16 [B,H,D,S].
// ---------------------------------------------------------------------------
#define XLD 40
#define WLD 136

__device__ __forceinline__ void stage_tiles(
    const bf16_t* __restrict__ X, const bf16_t* __restrict__ Wt,
    bf16_t* xT, bf16_t* wT, int m0blk, int n0blk, int k0, int K, int N,
    int t) {
  // X tile: 64 rows x 32 cols, 4 x 16B chunks per row, 256 chunks total
  int xr = t >> 2, xc = t & 3;
  async_copy16(X + (size_t)(m0blk + xr) * K + k0 + xc * 8,
               xT + xr * XLD + xc * 8);
  // W tile: 32 rows x 128 cols, 16 x 16B chunks per row, 512 chunks total
#pragma unroll
  for (int i = 0; i < 2; ++i) {
    int c = t + 256 * i;
    int wr = c >> 4, wc = c & 15;
    async_copy16(Wt + (size_t)(k0 + wr) * N + n0blk + wc * 8,
                 wT + wr * WLD + wc * 8);
  }
}

__global__ __launch_bounds__(256) void gemm_bf16_wmma(
    const bf16_t* __restrict__ X, const bf16_t* __restrict__ Wt,
    const float* __restrict__ bias, float* __restrict__ outF,
    bf16_t* __restrict__ outH, int M, int N, int K, int mode) {
  __shared__ __align__(16) bf16_t xTile[2][64 * XLD];
  __shared__ __align__(16) bf16_t wTile[2][32 * WLD];

  int t = threadIdx.x;
  int lane = t & 31;
  int wave = t >> 5;            // 0..7
  int wm = wave & 3, wn = wave >> 2;
  int m0blk = blockIdx.x * 64;
  int n0blk = blockIdx.y * 128;

  v8f acc[4];
#pragma unroll
  for (int q = 0; q < 4; ++q) acc[q] = (v8f){};

  stage_tiles(X, Wt, xTile[0], wTile[0], m0blk, n0blk, 0, K, N, t);

  int buf = 0;
  for (int k0 = 0; k0 < K; k0 += 32) {
    if (k0 + 32 < K) {
      stage_tiles(X, Wt, xTile[buf ^ 1], wTile[buf ^ 1], m0blk, n0blk,
                  k0 + 32, K, N, t);
      WAIT_ASYNC(3);   // async loads complete in order: current stage landed
    } else {
      WAIT_ASYNC(0);
    }
    __syncthreads();

    v16bf a = load_A(xTile[buf] + wm * 16 * XLD, XLD, lane);
#pragma unroll
    for (int q = 0; q < 4; ++q) {
      v16bf b = load_B(wTile[buf] + wn * 64 + 16 * q, WLD, lane);
      acc[q] = wmma_bf16(a, b, acc[q]);
    }
    __syncthreads();
    buf ^= 1;
  }

  // C layout: lane half selects row block (+0/+8), vgpr r is row, col = lane&15
  int rowoff = (lane < 16) ? 0 : 8;
  int ncol = lane & 15;
  int m0 = m0blk + wm * 16;
  int n0 = n0blk + wn * 64;
#pragma unroll
  for (int q = 0; q < 4; ++q) {
    int n = n0 + 16 * q + ncol;
    float bv = bias[n];
#pragma unroll
    for (int r = 0; r < 8; ++r) {
      float v = acc[q][r] + bv;
      int m = m0 + r + rowoff;
      if (mode == 0) {
        outF[(size_t)m * N + n] = v;
      } else {
        int b = m >> 11, s = m & (SEQ - 1);   // M = B*S, S = 2048
        int h = n >> 6, d = n & (HDIM - 1);   // N = E,  D = 64
        if (mode == 1)
          outH[((((size_t)b * HEADS + h) * SEQ + s) * HDIM) + d] = (bf16_t)v;
        else
          outH[((((size_t)b * HEADS + h) * HDIM + d) * SEQ) + s] = (bf16_t)v;
      }
    }
  }
}

// ---------------------------------------------------------------------------
// Flash attention: per wave one 16-row Q tile; stream 32 keys/step.
// Q [B,H,S,D] bf16, Kt [B,H,D,S] bf16, V [B,H,S,D] bf16, mask [B,S] bool.
// Output attn [B,S,E] bf16 with E index = h*64+d.
// ---------------------------------------------------------------------------
__global__ __launch_bounds__(128) void flash_attn_wmma(
    const bf16_t* __restrict__ Q, const bf16_t* __restrict__ Kt,
    const bf16_t* __restrict__ V, const unsigned char* __restrict__ mask,
    bf16_t* __restrict__ attn) {
  int lane = threadIdx.x & 31;
  int wave = threadIdx.x >> 5;   // 0..3
  int b = blockIdx.z, h = blockIdx.y;
  int q0 = blockIdx.x * 64 + wave * 16;

  const bf16_t* Qb  = Q  + (((size_t)b * HEADS + h) * SEQ) * HDIM;
  const bf16_t* Ktb = Kt + (((size_t)b * HEADS + h) * HDIM) * SEQ;
  const bf16_t* Vb  = V  + (((size_t)b * HEADS + h) * SEQ) * HDIM;
  const unsigned char* mb = mask + (size_t)b * SEQ;

  // per-wave LDS tile for P (C-layout -> A-layout transpose), 16x32, padded
  __shared__ __align__(16) bf16_t pLds[4][16][40];

  // Q tile fragments: d = 0..31 and 32..63
  v16bf aq0 = load_A(Qb + (size_t)q0 * HDIM, HDIM, lane);
  v16bf aq1 = load_A(Qb + (size_t)q0 * HDIM + 32, HDIM, lane);

  v8f o[4];
#pragma unroll
  for (int q = 0; q < 4; ++q) o[q] = (v8f){};
  float rowM[8], rowL[8];
#pragma unroll
  for (int r = 0; r < 8; ++r) { rowM[r] = -3.0e38f; rowL[r] = 0.0f; }

  const float scale = 0.125f;   // 1/sqrt(64)
  int rowoff = (lane < 16) ? 0 : 8;
  int ncol = lane & 15;

  for (int j0 = 0; j0 < SEQ; j0 += 32) {
    // scores: two 16x16 tiles over keys [j0, j0+32)
    v8f s0 = (v8f){}, s1 = (v8f){};
    s0 = wmma_bf16(aq0, load_B(Ktb + j0,                      SEQ, lane), s0);
    s0 = wmma_bf16(aq1, load_B(Ktb + (size_t)32 * SEQ + j0,      SEQ, lane), s0);
    s1 = wmma_bf16(aq0, load_B(Ktb + j0 + 16,                 SEQ, lane), s1);
    s1 = wmma_bf16(aq1, load_B(Ktb + (size_t)32 * SEQ + j0 + 16, SEQ, lane), s1);

    float pen0 = mb[j0 + ncol]      ? -1.0e30f : 0.0f;
    float pen1 = mb[j0 + 16 + ncol] ? -1.0e30f : 0.0f;

#pragma unroll
    for (int r = 0; r < 8; ++r) {
      float a0 = s0[r] * scale + pen0;
      float a1 = s1[r] * scale + pen1;
      // row max across the 16 lanes of this half (xor offsets stay in half)
      float mx = fmaxf(a0, a1);
#pragma unroll
      for (int off = 8; off >= 1; off >>= 1)
        mx = fmaxf(mx, __shfl_xor(mx, off, 32));
      float newM = fmaxf(rowM[r], mx);
      float alpha = __expf(rowM[r] - newM);
      rowM[r] = newM;
      float p0 = __expf(a0 - newM);
      float p1 = __expf(a1 - newM);
      float rs = p0 + p1;
#pragma unroll
      for (int off = 8; off >= 1; off >>= 1)
        rs += __shfl_xor(rs, off, 32);
      rowL[r] = rowL[r] * alpha + rs;
#pragma unroll
      for (int q = 0; q < 4; ++q) o[q][r] *= alpha;
      // stash P into LDS (row = r+rowoff, col = key offset)
      pLds[wave][r + rowoff][ncol]      = (bf16_t)p0;
      pLds[wave][r + rowoff][ncol + 16] = (bf16_t)p1;
    }

    // P (16x32) as A fragment from LDS; V rows are B fragments (k = key, lane&15 = d)
    v16bf ap = load_A(&pLds[wave][0][0], 40, lane);
#pragma unroll
    for (int q = 0; q < 4; ++q) {
      v16bf bv = load_B(Vb + (size_t)j0 * HDIM + 16 * q, HDIM, lane);
      o[q] = wmma_bf16(ap, bv, o[q]);
    }
  }

  // finalize: O /= L, store to attn [B,S,E] at E = h*64 + d
#pragma unroll
  for (int r = 0; r < 8; ++r) {
    float inv = 1.0f / rowL[r];
    int m = q0 + r + rowoff;
#pragma unroll
    for (int q = 0; q < 4; ++q) {
      int d = 16 * q + ncol;
      attn[((size_t)b * SEQ + m) * EMBED + h * HDIM + d] =
          (bf16_t)(o[q][r] * inv);
    }
  }
}

// ---------------------------------------------------------------------------
// Host side
// ---------------------------------------------------------------------------
extern "C" void kernel_launch(void* const* d_in, const int* in_sizes, int n_in,
                              void* d_out, int out_size, void* d_ws,
                              size_t ws_size, hipStream_t stream) {
  const float* q  = (const float*)d_in[0];
  const float* k  = (const float*)d_in[1];
  const float* v  = (const float*)d_in[2];
  const unsigned char* mask = (const unsigned char*)d_in[3];
  const float* Wq = (const float*)d_in[4];
  const float* bq = (const float*)d_in[5];
  const float* Wk = (const float*)d_in[6];
  const float* bk = (const float*)d_in[7];
  const float* Wv = (const float*)d_in[8];
  const float* bv = (const float*)d_in[9];
  const float* Wo = (const float*)d_in[10];
  const float* bo = (const float*)d_in[11];

  bf16_t* ws = (bf16_t*)d_ws;
  bf16_t* Xq  = ws;             // [M,E] bf16
  bf16_t* Xk  = Xq + BSE;
  bf16_t* Xv  = Xk + BSE;
  bf16_t* Wqt = Xv + BSE;       // [E,E] bf16 transposed
  bf16_t* Wkt = Wqt + EE;
  bf16_t* Wvt = Wkt + EE;
  bf16_t* Wot = Wvt + EE;
  bf16_t* Qh  = Wot + EE;       // [B,H,S,D]
  bf16_t* KtB = Qh + BSE;       // [B,H,D,S]
  bf16_t* Vh  = KtB + BSE;      // [B,H,S,D]
  bf16_t* At  = Vh + BSE;       // [B,S,E]

  // 1) casts
  int castBlocks = (int)((BSE + 255) / 256);
  cast_f32_to_bf16<<<castBlocks, 256, 0, stream>>>(q, Xq, BSE);
  cast_f32_to_bf16<<<castBlocks, 256, 0, stream>>>(k, Xk, BSE);
  cast_f32_to_bf16<<<castBlocks, 256, 0, stream>>>(v, Xv, BSE);
  dim3 tgrid(EMBED / 32, EMBED / 32);
  transpose_cast_w<<<tgrid, 256, 0, stream>>>(Wq, Wqt);
  transpose_cast_w<<<tgrid, 256, 0, stream>>>(Wk, Wkt);
  transpose_cast_w<<<tgrid, 256, 0, stream>>>(Wv, Wvt);
  transpose_cast_w<<<tgrid, 256, 0, stream>>>(Wo, Wot);

  // 2) QKV projections (WMMA GEMM, async double-buffered LDS staging)
  dim3 ggrid(MROWS / 64, EMBED / 128);
  gemm_bf16_wmma<<<ggrid, 256, 0, stream>>>(Xq, Wqt, bq, nullptr, Qh,
                                            MROWS, EMBED, EMBED, 1);
  gemm_bf16_wmma<<<ggrid, 256, 0, stream>>>(Xk, Wkt, bk, nullptr, KtB,
                                            MROWS, EMBED, EMBED, 2);
  gemm_bf16_wmma<<<ggrid, 256, 0, stream>>>(Xv, Wvt, bv, nullptr, Vh,
                                            MROWS, EMBED, EMBED, 1);

  // 3) flash attention
  dim3 fgrid(SEQ / 64, HEADS, BATCH);
  flash_attn_wmma<<<fgrid, 128, 0, stream>>>(Qh, KtB, Vh, mask, At);

  // 4) output projection -> fp32 d_out
  gemm_bf16_wmma<<<ggrid, 256, 0, stream>>>(At, Wot, bo, (float*)d_out,
                                            nullptr, MROWS, EMBED, EMBED, 0);
}
